// GNNGuard_38628935860962
// MI455X (gfx1250) — matrix operand order repaired and checked by reading, not compiled, not run on hardware
//
#include <hip/hip_runtime.h>
#include <hip/hip_bf16.h>
#include <cstdint>

// ---------------------------------------------------------------------------
// GNNGuard (2-layer GCN) for MI455X / gfx1250.
// Pipeline:
//   zero -> degree histogram -> exclusive scan (CSR offsets) -> bucket fill
//   -> WMMA GEMM1 (bf16x3 split, fp32 accum) -> gather-aggregate + bias + relu
//   -> WMMA GEMM2 -> gather-aggregate + bias -> log_softmax
// ---------------------------------------------------------------------------

#define NFEAT 512
#define NHID  128
#define NCLS  16

typedef __attribute__((ext_vector_type(16))) __bf16 v16bf;
typedef __attribute__((ext_vector_type(8)))  float  v8f;

// ----------------------------- utility kernels -----------------------------

__global__ void zero_i32_kernel(int* p, int n) {
  int i = blockIdx.x * blockDim.x + threadIdx.x;
  if (i < n) p[i] = 0;
}

__global__ void deg_kernel(const int* __restrict__ dst, int* __restrict__ deg, int E) {
  int e = blockIdx.x * blockDim.x + threadIdx.x;
  if (e < E) atomicAdd(&deg[dst[e]], 1);
}

// Single-block chunked exclusive scan: deg[0..n) -> exclusive offsets in place,
// cursor[i] = offsets[i], deg[n] = total.
__global__ __launch_bounds__(1024)
void scan_kernel(int* __restrict__ deg, int* __restrict__ cursor, int n) {
  __shared__ int buf[1024];
  __shared__ int carry_s;
  const int t = threadIdx.x;
  if (t == 0) carry_s = 0;
  __syncthreads();
  for (int base = 0; base < n; base += 1024) {
    const int i = base + t;
    const int v = (i < n) ? deg[i] : 0;
    buf[t] = v;
    __syncthreads();
    #pragma unroll
    for (int off = 1; off < 1024; off <<= 1) {
      int x = (t >= off) ? buf[t - off] : 0;
      __syncthreads();
      buf[t] += x;
      __syncthreads();
    }
    const int carry = carry_s;
    const int excl  = buf[t] - v + carry;
    if (i < n) { deg[i] = excl; cursor[i] = excl; }
    const int total = buf[1023];
    __syncthreads();
    if (t == 0) carry_s = carry + total;
    __syncthreads();
  }
  if (t == 0) deg[n] = carry_s;
}

__global__ void bucket_kernel(const int* __restrict__ src, const int* __restrict__ dst,
                              const float* __restrict__ w, int* __restrict__ cursor,
                              int* __restrict__ srcS, float* __restrict__ wS, int E) {
  int e = blockIdx.x * blockDim.x + threadIdx.x;
  if (e >= E) return;
  int p = atomicAdd(&cursor[dst[e]], 1);
  srcS[p] = src[e];
  wS[p]   = w[e];
}

// ------------------------------- WMMA GEMM ---------------------------------
// C[nrows x NOUT] = A[nrows x K] * B[K x NOUT], fp32 in/out.
// bf16x3 split: A = Ahi + Alo, B = Bhi + Blo; C += Ahi*Bhi + Ahi*Blo + Alo*Bhi.
// One 16-row stripe per block; wave w owns output columns [16w, 16w+16).
// A tiles (16x32 f32) staged into LDS via gfx1250 async global->LDS loads.
template<int K, int NOUT>
__global__ __launch_bounds__(32 * (NOUT / 16))
void wmma_gemm_bf16x3(const float* __restrict__ A, const float* __restrict__ B,
                      float* __restrict__ C, int nrows) {
  __shared__ float tileA[16 * 32];
  const int tid  = threadIdx.x;
  const int wave = tid >> 5;
  const int lane = tid & 31;
  const int half = lane >> 4;   // 0: lanes 0-15, 1: lanes 16-31
  const int idx  = lane & 15;
  const int rowbase = blockIdx.x * 16;
  const int colbase = wave * 16;
  if (rowbase >= nrows) return;

  v8f c = {};

  for (int kb = 0; kb < K; kb += 32) {
    __syncthreads();
    if (wave == 0) {
      // Stage 16x32 fp32 A-tile: 128 16-byte chunks, 4 async b128 ops per lane.
      #pragma unroll
      for (int j = 0; j < 4; ++j) {
        const int q  = j * 32 + lane;      // chunk id 0..127
        const int r  = q >> 3;             // row 0..15
        const int c4 = (q & 7) << 2;       // col 0,4,..,28
        const float* gp = A + (size_t)(rowbase + r) * K + kb + c4;
        unsigned lds_off = (unsigned)(uintptr_t)(&tileA[r * 32 + c4]);
        asm volatile("global_load_async_to_lds_b128 %0, %1, off"
                     :: "v"(lds_off), "v"(gp) : "memory");
      }
      asm volatile("s_wait_asynccnt 0x0" ::: "memory");
    }
    __syncthreads();

    // Build fragments per ISA 7.12.2 register layouts.
    v16bf a_hi, a_lo, b_hi, b_lo;
    #pragma unroll
    for (int e = 0; e < 16; ++e) {
      // A (16x32, 16-bit): element e of lane -> K = (e&8?16:0) + half*8 + (e&7)
      const int ka = ((e & 8) ? 16 : 0) + half * 8 + (e & 7);
      const float av = tileA[idx * 32 + ka];
      const __bf16 ah = (__bf16)av;
      a_hi[e] = ah;
      a_lo[e] = (__bf16)(av - (float)ah);
      // B (32x16, 16-bit): same K order as A -> K = (e&8?16:0) + half*8 + (e&7)
      const int kbb = ((e & 8) ? 16 : 0) + half * 8 + (e & 7);
      const float bv = B[(size_t)(kb + kbb) * NOUT + colbase + idx];
      const __bf16 bh = (__bf16)bv;
      b_hi[e] = bh;
      b_lo[e] = (__bf16)(bv - (float)bh);
    }
    c = __builtin_amdgcn_wmma_f32_16x16x32_bf16(false, a_hi, false, b_hi, (short)0, c, false, false);
    c = __builtin_amdgcn_wmma_f32_16x16x32_bf16(false, a_hi, false, b_lo, (short)0, c, false, false);
    c = __builtin_amdgcn_wmma_f32_16x16x32_bf16(false, a_lo, false, b_hi, (short)0, c, false, false);
  }

  // C/D layout: VGPR r -> row (rowbase + half*8 + r), column idx.
  #pragma unroll
  for (int r = 0; r < 8; ++r) {
    const int row = rowbase + half * 8 + r;
    if (row < nrows) C[(size_t)row * NOUT + colbase + idx] = c[r];
  }
}

// --------------------------- gather aggregation ----------------------------
// Layer 1: one wave per dst node; lane owns 4 of 128 features. No atomics.
__global__ __launch_bounds__(256)
void agg1_kernel(const float* __restrict__ t1, const int* __restrict__ off,
                 const int* __restrict__ srcS, const float* __restrict__ wS,
                 const float* __restrict__ b1, float* __restrict__ h, int n) {
  const int gw   = (blockIdx.x * 256 + threadIdx.x) >> 5;
  const int lane = threadIdx.x & 31;
  if (gw >= n) return;
  const int s = off[gw], e = off[gw + 1];
  const int f = lane * 4;
  float a0 = 0.f, a1 = 0.f, a2 = 0.f, a3 = 0.f;
  for (int j = s; j < e; ++j) {
    const int   sn = srcS[j];
    const float w  = wS[j];
    const float4 v = *(const float4*)(t1 + (size_t)sn * NHID + f);
    a0 += w * v.x; a1 += w * v.y; a2 += w * v.z; a3 += w * v.w;
  }
  float4 o;
  o.x = fmaxf(a0 + b1[f + 0], 0.f);
  o.y = fmaxf(a1 + b1[f + 1], 0.f);
  o.z = fmaxf(a2 + b1[f + 2], 0.f);
  o.w = fmaxf(a3 + b1[f + 3], 0.f);
  *(float4*)(h + (size_t)gw * NHID + f) = o;
}

// Layer 2: one thread per (node, class).
__global__ void agg2_kernel(const float* __restrict__ t2, const int* __restrict__ off,
                            const int* __restrict__ srcS, const float* __restrict__ wS,
                            const float* __restrict__ b2, float* __restrict__ agg, int n) {
  const int gid = blockIdx.x * blockDim.x + threadIdx.x;
  const int node = gid >> 4;
  const int f    = gid & 15;
  if (node >= n) return;
  const int s = off[node], e = off[node + 1];
  float acc = 0.f;
  for (int j = s; j < e; ++j) acc += wS[j] * t2[(size_t)srcS[j] * NCLS + f];
  agg[(size_t)node * NCLS + f] = acc + b2[f];
}

__global__ void logsoftmax_kernel(const float* __restrict__ agg, float* __restrict__ out, int n) {
  const int node = blockIdx.x * blockDim.x + threadIdx.x;
  if (node >= n) return;
  float v[NCLS];
  float m = -3.402823466e38f;
  #pragma unroll
  for (int c = 0; c < NCLS; ++c) { v[c] = agg[(size_t)node * NCLS + c]; m = fmaxf(m, v[c]); }
  float s = 0.f;
  #pragma unroll
  for (int c = 0; c < NCLS; ++c) s += __expf(v[c] - m);
  const float ls = __logf(s);
  #pragma unroll
  for (int c = 0; c < NCLS; ++c) out[(size_t)node * NCLS + c] = v[c] - m - ls;
}

// --------------------------------- launch ----------------------------------

static inline size_t align16(size_t x) { return (x + 15) & ~(size_t)15; }

extern "C" void kernel_launch(void* const* d_in, const int* in_sizes, int n_in,
                              void* d_out, int out_size, void* d_ws, size_t ws_size,
                              hipStream_t stream) {
  const float* x  = (const float*)d_in[0];
  const int*   ei = (const int*)d_in[1];
  const float* ew = (const float*)d_in[2];
  const float* W1 = (const float*)d_in[3];
  const float* b1 = (const float*)d_in[4];
  const float* W2 = (const float*)d_in[5];
  const float* b2 = (const float*)d_in[6];
  float* out = (float*)d_out;

  const int N = in_sizes[0] / NFEAT;   // 100000
  const int E = in_sizes[2];           // 1600000
  const int* src = ei;
  const int* dst = ei + E;

  // Workspace carve-up (all regions 16B aligned).
  char* w = (char*)d_ws;
  int*   off    = (int*)w;   w += align16((size_t)(N + 1) * 4);
  int*   cursor = (int*)w;   w += align16((size_t)N * 4);
  int*   srcS   = (int*)w;   w += align16((size_t)E * 4);
  float* wS     = (float*)w; w += align16((size_t)E * 4);
  float* t1     = (float*)w; w += align16((size_t)N * NHID * 4);
  float* h      = (float*)w; w += align16((size_t)N * NHID * 4);
  float* t2     = t1;                       // t1 dead after agg1
  float* agg2   = t1 + (size_t)N * NCLS;    // disjoint from t2 within t1 region
  (void)ws_size; (void)n_in; (void)out_size;

  // 1) CSR build
  zero_i32_kernel<<<(N + 1 + 255) / 256, 256, 0, stream>>>(off, N + 1);
  deg_kernel<<<(E + 255) / 256, 256, 0, stream>>>(dst, off, E);
  scan_kernel<<<1, 1024, 0, stream>>>(off, cursor, N);
  bucket_kernel<<<(E + 255) / 256, 256, 0, stream>>>(src, dst, ew, cursor, srcS, wS, E);

  // 2) Layer 1: GEMM (WMMA bf16x3) + gather-aggregate + bias + relu
  const int rowTiles = (N + 15) / 16;
  wmma_gemm_bf16x3<NFEAT, NHID><<<rowTiles, 32 * (NHID / 16), 0, stream>>>(x, W1, t1, N);
  agg1_kernel<<<(N * 32 + 255) / 256, 256, 0, stream>>>(t1, off, srcS, wS, b1, h, N);

  // 3) Layer 2: GEMM (WMMA bf16x3) + gather-aggregate + bias
  wmma_gemm_bf16x3<NHID, NCLS><<<rowTiles, 32 * (NCLS / 16), 0, stream>>>(h, W2, t2, N);
  agg2_kernel<<<(N * NCLS + 255) / 256, 256, 0, stream>>>(t2, off, srcS, wS, b2, agg2, N);

  // 4) log_softmax
  logsoftmax_kernel<<<(N + 255) / 256, 256, 0, stream>>>(agg2, out, N);
}
